// CrossAtt_51745765983009
// MI455X (gfx1250) — matrix operand order repaired and checked by reading, not compiled, not run on hardware
//
#include <hip/hip_runtime.h>
#include <hip/hip_bf16.h>

typedef __attribute__((ext_vector_type(16))) __bf16 v16bf;
typedef __attribute__((ext_vector_type(8)))  __bf16 v8bf;
typedef __attribute__((ext_vector_type(8)))  float  v8f;

#define N_P 128
#define N_S 512
#define DIM 128
#define NB  512
#define SCALE 0.08838834764831845f

__device__ __forceinline__ v8f wmma_bf16(v16bf a, v16bf b, v8f c) {
  return __builtin_amdgcn_wmma_f32_16x16x32_bf16(false, a, false, b, (short)0, c,
                                                 false, false);
}

// CDNA5 async global->LDS copy, 16B per lane (tracked by ASYNCcnt).
__device__ __forceinline__ void async_g2l_16B(unsigned lds_addr, const void* gaddr) {
  asm volatile("global_load_async_to_lds_b128 %0, %1, off"
               :
               : "v"(lds_addr), "v"(gaddr)
               : "memory");
}
__device__ __forceinline__ void wait_async0() {
  asm volatile("s_wait_asynccnt 0" ::: "memory");
}

// A-fragment (16x32 bf16) from a row-major tile: lane holds row M=lane%16.
// elements e<8  -> K = k0 + 16*half + e        (contiguous 8)
// elements e>=8 -> K = k0 + 16*half + 8 + e-8  (contiguous 8)
__device__ __forceinline__ v16bf make_afrag(const __bf16* row, int k0, int half) {
  const v8bf lo = *(const v8bf*)(row + k0 + 16 * half);
  const v8bf hi = *(const v8bf*)(row + k0 + 16 * half + 8);
  v16bf a;
#pragma unroll
  for (int e = 0; e < 8; ++e) { a[e] = lo[e]; a[e + 8] = hi[e]; }
  return a;
}

// One 16-row x 128-col GEMM stripe from LDS tiles (8 live accumulators).
__device__ __forceinline__ void gemm_stripe(v8f acc[8], const __bf16* Xrow,
                                            const __bf16* Wt, int m, int half) {
#pragma unroll
  for (int t = 0; t < 8; ++t) acc[t] = (v8f){};
#pragma unroll
  for (int k0 = 0; k0 < 128; k0 += 32) {
    v16bf a = make_afrag(Xrow, k0, half);
#pragma unroll
    for (int nt = 0; nt < 8; ++nt) {
      v16bf b = *(const v16bf*)(Wt + (nt * 16 + m) * 128 + k0 + 16 * half);
      acc[nt] = wmma_bf16(a, b, acc[nt]);
    }
  }
}

// ---------------------------------------------------------------------------
// Kernel 1: O1[M x 128] = bf16( X[M x 128] @ W1 )   (row-major)
// DUAL: additionally O2t[b][d][s] = bf16( X @ W2 ) transposed per batch of
// 512 rows, so kernel 2 can async-DMA V tiles straight into LDS.
// Grid: M/128 workgroups of 256 threads (8 waves, 16 rows each).
// Dynamic LDS: X bf16 32KB + W1^T 32KB (+ W2^T 32KB when DUAL).
// ---------------------------------------------------------------------------
template <bool DUAL>
__global__ __launch_bounds__(256) void qkv_gemm(const float* __restrict__ X,
                                                const float* __restrict__ W1,
                                                const float* __restrict__ W2,
                                                __bf16* __restrict__ O1,
                                                __bf16* __restrict__ O2t) {
  extern __shared__ char smem[];
  __bf16* Xl  = (__bf16*)smem;             // 128x128 row-major (reused for V^T)
  __bf16* W1t = (__bf16*)(smem + 32768);   // [o][h] transposed
  __bf16* W2t = (__bf16*)(smem + 65536);
  const int tid = threadIdx.x;
  const long row0 = (long)blockIdx.x * 128;

  for (int i = tid; i < 128 * 128; i += 256) Xl[i] = (__bf16)X[row0 * 128 + i];
  for (int i = tid; i < 128 * 128; i += 256) {
    int h = i >> 7, o = i & 127;
    W1t[o * 128 + h] = (__bf16)W1[i];
  }
  if (DUAL)
    for (int i = tid; i < 128 * 128; i += 256) {
      int h = i >> 7, o = i & 127;
      W2t[o * 128 + h] = (__bf16)W2[i];
    }
  __syncthreads();

  const int wave = tid >> 5, lane = tid & 31;
  const int m = lane & 15, half = lane >> 4;
  const __bf16* Xrow = Xl + (wave * 16 + m) * 128;

  {  // pass 1: O1 row-major
    v8f acc[8];
    gemm_stripe(acc, Xrow, W1t, m, half);
#pragma unroll
    for (int nt = 0; nt < 8; ++nt)
#pragma unroll
      for (int r = 0; r < 8; ++r)
        O1[(row0 + wave * 16 + r + 8 * half) * 128 + nt * 16 + m] =
            (__bf16)acc[nt][r];
  }

  if (DUAL) {  // pass 2: O2 transposed (via LDS, coalesced 16B stores)
    v8f acc[8];
    gemm_stripe(acc, Xrow, W2t, m, half);
    __syncthreads();  // all waves done reading Xl
#pragma unroll
    for (int nt = 0; nt < 8; ++nt)
#pragma unroll
      for (int r = 0; r < 8; ++r)  // XlT[d][s_in_block]
        Xl[(nt * 16 + m) * 128 + wave * 16 + r + 8 * half] = (__bf16)acc[nt][r];
    __syncthreads();
    const int bb = (int)(row0 >> 9);       // batch
    const int s0 = (int)(row0 & 511);      // s offset within batch
    for (int j = tid; j < 2048; j += 256) {  // 128 rows x 256B
      int d = j >> 4, off8 = (j & 15) * 8;
      *(uint4*)(O2t + (long)bb * (512 * 128) + d * 512 + s0 + off8) =
          *(const uint4*)(Xl + d * 128 + off8);
    }
  }
}

// ---------------------------------------------------------------------------
// Kernel 2: one workgroup (256 thr, 8 waves) per batch.
// LDS layout (dynamic, 316416 B):
//   e    f32 [128][512]  @ 0        (256KB)  (reused for Wfc^T bf16 at the end)
//   q    bf16 [128][128] @ 262144   (32KB)   (reused for h_sel bf16)
//   vt0  bf16 [128][32]  @ 294912   (8KB)    async double-buffered V^T tiles
//   vt1  bf16 [128][32]  @ 303104   (8KB)
//   ppos f32 [128][2]    @ 311296   (1KB)
//   spos f32 [512][2]    @ 312320   (4KB)
// ---------------------------------------------------------------------------
__global__ __launch_bounds__(256) void attn_kernel(
    const float* __restrict__ obs, const float* __restrict__ p_hidden,
    const __bf16* __restrict__ Qw, const __bf16* __restrict__ Kw,
    const __bf16* __restrict__ Vt, const float* __restrict__ Wfc,
    const float* __restrict__ bfc, float* __restrict__ hout,
    float* __restrict__ alpha) {
  extern __shared__ char smem[];
  float*  e_lds = (float*)smem;
  __bf16* q_lds = (__bf16*)(smem + 262144);
  __bf16* vt0   = (__bf16*)(smem + 294912);
  __bf16* vt1   = (__bf16*)(smem + 303104);
  float*  ppos  = (float*)(smem + 311296);
  float*  spos  = (float*)(smem + 312320);
  __bf16* wfct  = (__bf16*)smem;  // reuse of e region after pass 2

  const int b = blockIdx.x;
  const int tid = threadIdx.x;
  const __bf16* Kb  = Kw + (long)b * N_S * DIM;
  const __bf16* Vtb = Vt + (long)b * N_S * DIM;  // [d][s] layout

  // Async prefetch of V^T chunk 0 (128 x 32) — streams in under pass 1.
  {
    int j0 = tid, j1 = tid + 256;  // 512 x 16B transfers
    async_g2l_16B((unsigned)(uintptr_t)(vt0 + (j0 >> 2) * 32 + (j0 & 3) * 8),
                  Vtb + (j0 >> 2) * 512 + (j0 & 3) * 8);
    async_g2l_16B((unsigned)(uintptr_t)(vt0 + (j1 >> 2) * 32 + (j1 & 3) * 8),
                  Vtb + (j1 >> 2) * 512 + (j1 & 3) * 8);
  }

  {  // stage q (bf16, 32KB) as 16B chunks
    const uint4* src = (const uint4*)(Qw + (long)b * N_P * DIM);
    uint4* dst = (uint4*)q_lds;
    for (int i = tid; i < 2048; i += 256) dst[i] = src[i];
  }
  for (int i = tid; i < N_P; i += 256) {
    ppos[i * 2 + 0] = obs[((long)b * (N_P + N_S) + i) * 4 + 0];
    ppos[i * 2 + 1] = obs[((long)b * (N_P + N_S) + i) * 4 + 1];
  }
  for (int i = tid; i < N_S; i += 256) {
    spos[i * 2 + 0] = obs[((long)b * (N_P + N_S) + N_P + i) * 4 + 0];
    spos[i * 2 + 1] = obs[((long)b * (N_P + N_S) + N_P + i) * 4 + 1];
  }
  __syncthreads();

  const int wave = tid >> 5, lane = tid & 31;
  const int m = lane & 15, half = lane >> 4;
  const int m0 = wave * 16;

  // Hoist this lane's 8 predator positions into registers: they are invariant
  // across all 32 s-tiles (compiler can't hoist past the e_lds stores itself).
  float pxr[8], pyr[8];
#pragma unroll
  for (int r = 0; r < 8; ++r) {
    int p = m0 + r + 8 * half;
    pxr[r] = ppos[p * 2 + 0];
    pyr[r] = ppos[p * 2 + 1];
  }

  // ---- pass 1: e = mask(q @ k^T * scale) into LDS -------------------------
  for (int st = 0; st < 32; ++st) {
    const int n0 = st * 16;
    if (st + 1 < 32)  // hide K fragment latency (global_prefetch_b8)
      __builtin_prefetch(Kb + (long)(n0 + 16 + m) * 128, 0, 0);
    v8f acc = (v8f){};
#pragma unroll
    for (int k0 = 0; k0 < 128; k0 += 32) {
      v16bf a  = make_afrag(q_lds + (m0 + m) * 128, k0, half);
      // B col n = K row (n0+n): contiguous in d -> direct 32B load
      v16bf bk = *(const v16bf*)(Kb + (long)(n0 + m) * 128 + k0 + 16 * half);
      acc = wmma_bf16(a, bk, acc);
    }
    const float sx = spos[(n0 + m) * 2], sy = spos[(n0 + m) * 2 + 1];
#pragma unroll
    for (int r = 0; r < 8; ++r) {
      int p = m0 + r + 8 * half;
      float dx = fabsf(pxr[r] - sx), dy = fabsf(pyr[r] - sy);
      float ev = (fmaxf(dx, dy) <= 0.2f) ? acc[r] * SCALE : -1.0e9f;
      e_lds[p * 512 + n0 + m] = ev;
    }
  }

  // ---- per-row max / sum (lane L owns row m0 + L%16, half the columns) ----
  float rmax, rsum;
  {
    const float4* rp = (const float4*)(e_lds + (m0 + m) * 512 + half * 256);
    float mx = -3.0e38f;
    for (int j = 0; j < 64; ++j) {
      float4 v = rp[j];
      mx = fmaxf(mx, fmaxf(fmaxf(v.x, v.y), fmaxf(v.z, v.w)));
    }
    mx = fmaxf(mx, __shfl_xor(mx, 16, 32));
    float sm = 0.f;
    for (int j = 0; j < 64; ++j) {
      float4 v = rp[j];
      sm += __expf(v.x - mx) + __expf(v.y - mx) + __expf(v.z - mx) + __expf(v.w - mx);
    }
    sm += __shfl_xor(sm, 16, 32);
    rmax = mx; rsum = sm;
  }
  const bool anynb = rmax > -1.0e8f;
  const float inv_s = anynb ? (1.0f / rsum) : 0.0f;  // !any_nb => alpha = 0

  // ---- pass 2: alpha (written out) and h_att = alpha @ V ------------------
  float* arow = alpha + ((long)b * N_P + m0 + m) * N_S;
  v8f h_acc[8];
#pragma unroll
  for (int nt = 0; nt < 8; ++nt) h_acc[nt] = (v8f){};

  for (int c = 0; c < 16; ++c) {
    __bf16* cur = (c & 1) ? vt1 : vt0;
    __bf16* nxt = (c & 1) ? vt0 : vt1;
    wait_async0();    // my async transfers for chunk c landed
    __syncthreads();  // everyone's landed; everyone done with `nxt` buffer
    if (c + 1 < 16) {  // async prefetch chunk c+1 while computing chunk c
      int j0 = tid, j1 = tid + 256;
      async_g2l_16B((unsigned)(uintptr_t)(nxt + (j0 >> 2) * 32 + (j0 & 3) * 8),
                    Vtb + (j0 >> 2) * 512 + (c + 1) * 32 + (j0 & 3) * 8);
      async_g2l_16B((unsigned)(uintptr_t)(nxt + (j1 >> 2) * 32 + (j1 & 3) * 8),
                    Vtb + (j1 >> 2) * 512 + (c + 1) * 32 + (j1 & 3) * 8);
    }

    // alpha values in A-fragment lane layout (lane owns row m0+m)
    const float* er = e_lds + (m0 + m) * 512 + c * 32;
    float4 v0 = *(const float4*)(er + 16 * half);
    float4 v1 = *(const float4*)(er + 16 * half + 4);
    float4 v2 = *(const float4*)(er + 16 * half + 8);
    float4 v3 = *(const float4*)(er + 16 * half + 12);
    float av[16] = {v0.x, v0.y, v0.z, v0.w, v1.x, v1.y, v1.z, v1.w,
                    v2.x, v2.y, v2.z, v2.w, v3.x, v3.y, v3.z, v3.w};
    v16bf afrag;
#pragma unroll
    for (int e = 0; e < 16; ++e) {
      float a = __expf(av[e] - rmax) * inv_s;  // masked e=-1e9 -> exp underflows to 0
      av[e] = a;
      afrag[e] = (__bf16)a;
    }
    *(float4*)(arow + c * 32 + 16 * half)          = make_float4(av[0], av[1], av[2], av[3]);
    *(float4*)(arow + c * 32 + 16 * half + 4)      = make_float4(av[4], av[5], av[6], av[7]);
    *(float4*)(arow + c * 32 + 16 * half + 8)      = make_float4(av[8], av[9], av[10], av[11]);
    *(float4*)(arow + c * 32 + 16 * half + 12)     = make_float4(av[12], av[13], av[14], av[15]);

#pragma unroll
    for (int nt = 0; nt < 8; ++nt) {
      v16bf bv = *(const v16bf*)(cur + (nt * 16 + m) * 32 + 16 * half);
      h_acc[nt] = wmma_bf16(afrag, bv, h_acc[nt]);
    }
  }

  // ---- h_sel = any_nb ? h_att : ph  (stored bf16 into reused q region) ----
  unsigned long long amask = __ballot(anynb);  // bit L = any_nb(row m0 + L%16)
#pragma unroll
  for (int nt = 0; nt < 8; ++nt) {
#pragma unroll
    for (int r = 0; r < 8; ++r) {
      int M = r + 8 * half;
      float ph = p_hidden[((long)b * N_P + m0 + M) * DIM + nt * 16 + m];
      float hv = ((amask >> M) & 1ull) ? h_acc[nt][r] : ph;
      q_lds[(m0 + M) * 128 + nt * 16 + m] = (__bf16)hv;
    }
  }
  __syncthreads();
  for (int i = tid; i < 128 * 128; i += 256) {  // stage Wfc^T bf16 (reuses e region)
    int h = i >> 7, o = i & 127;
    wfct[o * 128 + h] = (__bf16)Wfc[i];
  }
  __syncthreads();

  // ---- h_out = h_sel @ Wfc + b --------------------------------------------
  v8f oacc[8];
#pragma unroll
  for (int nt = 0; nt < 8; ++nt) oacc[nt] = (v8f){};
#pragma unroll
  for (int k0 = 0; k0 < 128; k0 += 32) {
    v16bf a = make_afrag(q_lds + (m0 + m) * 128, k0, half);
#pragma unroll
    for (int nt = 0; nt < 8; ++nt) {
      v16bf bw = *(const v16bf*)(wfct + (nt * 16 + m) * 128 + k0 + 16 * half);
      oacc[nt] = wmma_bf16(a, bw, oacc[nt]);
    }
  }
#pragma unroll
  for (int nt = 0; nt < 8; ++nt) {
    float bias = bfc[nt * 16 + m];
#pragma unroll
    for (int r = 0; r < 8; ++r)
      hout[((long)b * N_P + m0 + r + 8 * half) * DIM + nt * 16 + m] =
          oacc[nt][r] + bias;
  }
}

extern "C" void kernel_launch(void* const* d_in, const int* in_sizes, int n_in,
                              void* d_out, int out_size, void* d_ws, size_t ws_size,
                              hipStream_t stream) {
  (void)in_sizes; (void)n_in; (void)out_size; (void)ws_size;
  const float* obs      = (const float*)d_in[0];
  const float* p_hidden = (const float*)d_in[1];
  const float* s_hidden = (const float*)d_in[2];
  // d_in[3] = batch_size (compile-time constant NB=512 used)
  const float* Wq  = (const float*)d_in[4];
  const float* Wk  = (const float*)d_in[5];
  const float* Wv  = (const float*)d_in[6];
  const float* Wfc = (const float*)d_in[7];
  const float* bfc = (const float*)d_in[8];

  float* hout  = (float*)d_out;
  float* alpha = (float*)d_out + (size_t)NB * N_P * DIM;

  char* ws = (char*)d_ws;
  __bf16* Qw = (__bf16*)ws;                                  // 16 MB, [row][d]
  __bf16* Kw = (__bf16*)(ws + (size_t)16 * 1024 * 1024);     // 64 MB, [row][d]
  __bf16* Vt = (__bf16*)(ws + (size_t)80 * 1024 * 1024);     // 64 MB, [b][d][s]

  // Q = p_hidden @ Wq            (M = B*N_P   = 65536  -> 512 WGs)
  qkv_gemm<false><<<(NB * N_P) / 128, 256, 65536, stream>>>(p_hidden, Wq, nullptr,
                                                            Qw, nullptr);
  // K,V = s_hidden @ {Wk,Wv}     (M = B*N_S   = 262144 -> 2048 WGs; V transposed)
  qkv_gemm<true><<<(NB * N_S) / 128, 256, 98304, stream>>>(s_hidden, Wk, Wv, Kw, Vt);
  // attention + alpha + FC, one WG per batch, ~309KB dynamic LDS
  attn_kernel<<<NB, 256, 316416, stream>>>(obs, p_hidden, Qw, Kw, Vt, Wfc, bfc,
                                           hout, alpha);
}
